// TrajectoryLSTMDecoder_45921790328942
// MI455X (gfx1250) — matrix operand order repaired and checked by reading, not compiled, not run on hardware
//
#include <hip/hip_runtime.h>
#include <math.h>

#define B_  16384
#define C_  512
#define H_  1024
#define IN_ 516
#define HG_ 4096      // 4*H
#define HE_ 1056      // H + 32  (padded K: +4 extra features, +28 zeros)
#define T_  60

typedef __attribute__((ext_vector_type(16))) __bf16 v16bf;
typedef __attribute__((ext_vector_type(8)))  __bf16 v8bf;
typedef __attribute__((ext_vector_type(8)))  float  v8f;

__device__ __forceinline__ float sigmoidf_(float x) { return 1.0f / (1.0f + expf(-x)); }

// ---------------------------------------------------------------------------
// f32 -> bf16 pack (strided source, packed dest)
// ---------------------------------------------------------------------------
__global__ __launch_bounds__(256)
void cvt_bf16_kernel(const float* __restrict__ src, int srcStride,
                     __bf16* __restrict__ dst, int rows, int cols) {
    int i = blockIdx.x * 256 + threadIdx.x;
    if (i >= rows * cols) return;
    int r = i / cols, c = i - r * cols;
    dst[(size_t)r * cols + c] = (__bf16)src[(size_t)r * srcStride + c];
}

// ---------------------------------------------------------------------------
// Build W_hh_ext [4H, HE] bf16 : cols [0,H) = W_hh, cols [H,H+4) = W_extra, rest 0
// ---------------------------------------------------------------------------
__global__ __launch_bounds__(256)
void build_whh_ext_kernel(const float* __restrict__ Whh, const float* __restrict__ Wih,
                          __bf16* __restrict__ dst) {
    int i = blockIdx.x * 256 + threadIdx.x;
    if (i >= HG_ * HE_) return;
    int j = i / HE_, k = i - j * HE_;
    float v = 0.0f;
    if (k < H_)            v = Whh[(size_t)j * H_ + k];
    else if (k < H_ + 4)   v = Wih[(size_t)j * IN_ + C_ + (k - H_)];
    dst[i] = (__bf16)v;
}

// ---------------------------------------------------------------------------
// init pos/delta state
// ---------------------------------------------------------------------------
__global__ __launch_bounds__(256)
void init_state_kernel(const float* __restrict__ lp, const float* __restrict__ ld,
                       float* __restrict__ pos, float* __restrict__ del) {
    int i = blockIdx.x * 256 + threadIdx.x;
    if (i < B_ * 2) { pos[i] = lp[i]; del[i] = ld[i]; }
}

// ---------------------------------------------------------------------------
// Fill padded feature columns of h_ext: (pd, heading, zeros)
// ---------------------------------------------------------------------------
__global__ __launch_bounds__(256)
void fill_extra_kernel(const float* __restrict__ deltaState, __bf16* __restrict__ h_ext) {
    int b = blockIdx.x * 256 + threadIdx.x;
    if (b >= B_) return;
    float dx = deltaState[b * 2 + 0], dy = deltaState[b * 2 + 1];
    float nrm = fmaxf(sqrtf(dx * dx + dy * dy), 1e-6f);
    __bf16* p = h_ext + (size_t)b * HE_ + H_;
    p[0] = (__bf16)dx; p[1] = (__bf16)dy;
    p[2] = (__bf16)(dx / nrm); p[3] = (__bf16)(dy / nrm);
#pragma unroll
    for (int i = 4; i < 32; ++i) p[i] = (__bf16)0.0f;
}

// ---------------------------------------------------------------------------
// WMMA bf16 GEMM:  Out[M,N] = act( A[M,K] * Bm[N,K]^T + bias )  (+ addsrc)
//   mode 0: tanh   mode 1: none   mode 2: + addsrc   mode 3: GELU
//   outF (f32, ld=N) and/or outBf (bf16, ld=ldoBf) written if non-null.
// Block tile 128x256, BK=32, 8 waves (2x4); wave tile 64x64 = 4x4 WMMA
// 16x16x32 bf16. Double-buffered LDS staged via GLOBAL_LOAD_ASYNC_TO_LDS_B128
// (ASYNCcnt pipeline: copy tile k+1 while computing tile k).
// ---------------------------------------------------------------------------
#define BM 128
#define BN 256
#define BK 32
#define LDSS 48   // padded LDS row stride (bf16 elems), 96B = 16B aligned

__global__ __launch_bounds__(256, 1)
void gemm_bf16_wmma(const __bf16* __restrict__ A, int lda,
                    const __bf16* __restrict__ Bm, int ldb,
                    const float* __restrict__ bias,
                    const float* __restrict__ addsrc,
                    float* __restrict__ outF,
                    __bf16* __restrict__ outBf, int ldoBf,
                    int N, int K, int mode) {
    __shared__ __bf16 As[2][BM * LDSS];   // 2 x 12 KB
    __shared__ __bf16 Bs[2][BN * LDSS];   // 2 x 24 KB

    const int tid   = threadIdx.x;
    const int lane  = tid & 31;
    const int wave  = tid >> 5;
    const int waveM = wave >> 2;         // 0..1  (64 rows each)
    const int waveN = wave & 3;          // 0..3  (64 cols each)
    const int g     = lane >> 4;         // lane half
    const int ln    = lane & 15;
    const size_t rowBase = (size_t)blockIdx.y * BM;
    const size_t colBase = (size_t)blockIdx.x * BN;

    // async memory -> LDS stage of one 128x32 A tile and one 256x32 B tile
    auto stage_async = [&](int buf, int kk) {
#pragma unroll
        for (int it = 0; it < 2; ++it) {          // A: 512 16B chunks / 256 thr
            int c = tid + it * 256;
            int r = c >> 2, q = c & 3;
            unsigned ldsOff = (unsigned)(unsigned long long)
                (const void*)&As[buf][r * LDSS + q * 8];
            const __bf16* gp = &A[(rowBase + r) * lda + kk + q * 8];
            asm volatile("global_load_async_to_lds_b128 %0, %1, off"
                         :: "v"(ldsOff), "v"(gp) : "memory");
        }
#pragma unroll
        for (int it = 0; it < 4; ++it) {          // B: 1024 16B chunks / 256 thr
            int c = tid + it * 256;
            int r = c >> 2, q = c & 3;
            unsigned ldsOff = (unsigned)(unsigned long long)
                (const void*)&Bs[buf][r * LDSS + q * 8];
            const __bf16* gp = &Bm[(colBase + r) * ldb + kk + q * 8];
            asm volatile("global_load_async_to_lds_b128 %0, %1, off"
                         :: "v"(ldsOff), "v"(gp) : "memory");
        }
    };

    v8f acc[4][4];
#pragma unroll
    for (int mi = 0; mi < 4; ++mi)
#pragma unroll
        for (int ni = 0; ni < 4; ++ni) {
            v8f z = {0.f, 0.f, 0.f, 0.f, 0.f, 0.f, 0.f, 0.f};
            acc[mi][ni] = z;
        }

    // prologue: fill buffer 0
    stage_async(0, 0);
    asm volatile("s_wait_asynccnt 0" ::: "memory");
    __syncthreads();

    int p = 0;
    for (int kk = 0; kk < K; kk += BK) {
        if (kk + BK < K) stage_async(p ^ 1, kk + BK);   // overlap with compute

        // B 32x16 frags: lane g*16+n holds K {16g..16g+15} of column n
        v16bf bfrag[4];
#pragma unroll
        for (int ni = 0; ni < 4; ++ni) {
            int br = waveN * 64 + ni * 16 + ln;
            v8bf lo = *(const v8bf*)&Bs[p][br * LDSS + 16 * g];
            v8bf hi = *(const v8bf*)&Bs[p][br * LDSS + 16 * g + 8];
            bfrag[ni] = __builtin_shufflevector(lo, hi, 0, 1, 2, 3, 4, 5, 6, 7,
                                                8, 9, 10, 11, 12, 13, 14, 15);
        }
#pragma unroll
        for (int mi = 0; mi < 4; ++mi) {
            // A 16x32 frag: lane g*16+m holds K {16g..16g+15} of row m
            int ar = waveM * 64 + mi * 16 + ln;
            v8bf lo = *(const v8bf*)&As[p][ar * LDSS + 16 * g];
            v8bf hi = *(const v8bf*)&As[p][ar * LDSS + 16 * g + 8];
            v16bf afrag = __builtin_shufflevector(lo, hi, 0, 1, 2, 3, 4, 5, 6, 7,
                                                  8, 9, 10, 11, 12, 13, 14, 15);
#pragma unroll
            for (int ni = 0; ni < 4; ++ni)
                acc[mi][ni] = __builtin_amdgcn_wmma_f32_16x16x32_bf16(
                    false, afrag, false, bfrag[ni],
                    (short)0, acc[mi][ni], false, false);
        }

        asm volatile("s_wait_asynccnt 0" ::: "memory");  // next buffer landed
        __syncthreads();                                  // all reads of p done
        p ^= 1;
    }

    // epilogue: lanes 0-15 -> N=lane, M=r ; lanes 16-31 -> N=lane-16, M=8+r
#pragma unroll
    for (int mi = 0; mi < 4; ++mi)
#pragma unroll
        for (int ni = 0; ni < 4; ++ni) {
#pragma unroll
            for (int r = 0; r < 8; ++r) {
                size_t row = rowBase + waveM * 64 + mi * 16 + g * 8 + r;
                size_t col = colBase + waveN * 64 + ni * 16 + ln;
                float v = acc[mi][ni][r] + bias[col];
                if (mode == 2)      v += addsrc[row * (size_t)N + col];
                else if (mode == 0) v = tanhf(v);
                else if (mode == 3) v = 0.5f * v * (1.0f + erff(v * 0.70710678118f));
                if (outF)  outF[row * (size_t)N + col] = v;
                if (outBf) outBf[row * (size_t)ldoBf + col] = (__bf16)v;
            }
        }
}

// ---------------------------------------------------------------------------
// LSTM pointwise + LayerNorm. One block (256 thr) per batch row.
// ---------------------------------------------------------------------------
__global__ __launch_bounds__(256)
void lstm_pointwise_ln(const float* __restrict__ gates, float* __restrict__ cstate,
                       __bf16* __restrict__ h_ext,
                       const float* __restrict__ ln_g, const float* __restrict__ ln_b,
                       __bf16* __restrict__ yln) {
    const int b = blockIdx.x;
    const size_t gbase = (size_t)b * HG_;
    const size_t hbase = (size_t)b * H_;
    float hv[4];
    float sum = 0.f, sq = 0.f;
#pragma unroll
    for (int k = 0; k < 4; ++k) {
        int j = threadIdx.x + k * 256;
        float gi = gates[gbase + j];
        float gf = gates[gbase + H_ + j];
        float gg = gates[gbase + 2 * H_ + j];
        float go = gates[gbase + 3 * H_ + j];
        float cN = sigmoidf_(gf) * cstate[hbase + j] + sigmoidf_(gi) * tanhf(gg);
        float hN = sigmoidf_(go) * tanhf(cN);
        cstate[hbase + j] = cN;
        h_ext[(size_t)b * HE_ + j] = (__bf16)hN;
        hv[k] = hN;
        sum += hN; sq += hN * hN;
    }
#pragma unroll
    for (int o = 16; o > 0; o >>= 1) {
        sum += __shfl_xor(sum, o, 32);
        sq  += __shfl_xor(sq,  o, 32);
    }
    __shared__ float s1[8], s2[8];
    if ((threadIdx.x & 31) == 0) { s1[threadIdx.x >> 5] = sum; s2[threadIdx.x >> 5] = sq; }
    __syncthreads();
    float tot = 0.f, tot2 = 0.f;
#pragma unroll
    for (int i = 0; i < 8; ++i) { tot += s1[i]; tot2 += s2[i]; }
    float mu = tot * (1.0f / H_);
    float var = tot2 * (1.0f / H_) - mu * mu;
    float rs = rsqrtf(var + 1e-5f);
#pragma unroll
    for (int k = 0; k < 4; ++k) {
        int j = threadIdx.x + k * 256;
        yln[hbase + j] = (__bf16)((hv[k] - mu) * rs * ln_g[j] + ln_b[j]);
    }
}

// ---------------------------------------------------------------------------
// Head: delta = y @ W2^T + b2 ; pos += delta ; out[b, t, :] = pos.
// One wave32 per row, 8 rows per block.
// ---------------------------------------------------------------------------
__global__ __launch_bounds__(256)
void head_kernel(const __bf16* __restrict__ y, const float* __restrict__ W2,
                 const float* __restrict__ b2, float* __restrict__ posState,
                 float* __restrict__ deltaState, float* __restrict__ out, int t) {
    const int row  = blockIdx.x * 8 + (threadIdx.x >> 5);
    const int lane = threadIdx.x & 31;
    float s0 = 0.f, s1 = 0.f;
    const size_t ybase = (size_t)row * H_;
    for (int j = lane; j < H_; j += 32) {
        float yv = (float)y[ybase + j];
        s0 += yv * W2[j];
        s1 += yv * W2[H_ + j];
    }
#pragma unroll
    for (int o = 16; o > 0; o >>= 1) {
        s0 += __shfl_xor(s0, o, 32);
        s1 += __shfl_xor(s1, o, 32);
    }
    if (lane == 0) {
        float d0 = s0 + b2[0], d1 = s1 + b2[1];
        float p0 = posState[row * 2 + 0] + d0;
        float p1 = posState[row * 2 + 1] + d1;
        posState[row * 2 + 0] = p0;  posState[row * 2 + 1] = p1;
        deltaState[row * 2 + 0] = d0; deltaState[row * 2 + 1] = d1;
        out[(size_t)row * (T_ * 2) + t * 2 + 0] = p0;
        out[(size_t)row * (T_ * 2) + t * 2 + 1] = p1;
    }
}

// ---------------------------------------------------------------------------
extern "C" void kernel_launch(void* const* d_in, const int* in_sizes, int n_in,
                              void* d_out, int out_size, void* d_ws, size_t ws_size,
                              hipStream_t stream) {
    (void)in_sizes; (void)n_in; (void)out_size; (void)ws_size;
    const float* context = (const float*)d_in[0];
    const float* last_pos   = (const float*)d_in[1];
    const float* last_delta = (const float*)d_in[2];
    const float* Wh   = (const float*)d_in[3];
    const float* bh   = (const float*)d_in[4];
    const float* Wc   = (const float*)d_in[5];
    const float* bc   = (const float*)d_in[6];
    const float* W_ih = (const float*)d_in[7];
    const float* b_ih = (const float*)d_in[8];
    const float* W_hh = (const float*)d_in[9];
    const float* b_hh = (const float*)d_in[10];
    const float* ln_g = (const float*)d_in[11];
    const float* ln_b = (const float*)d_in[12];
    const float* W1   = (const float*)d_in[13];
    const float* b1   = (const float*)d_in[14];
    const float* W2   = (const float*)d_in[15];
    const float* b2   = (const float*)d_in[16];
    float* out = (float*)d_out;

    // carve workspace
    char* w = (char*)d_ws;
    auto carve = [&](size_t bytes) -> void* {
        void* p = (void*)w;
        w += (bytes + 255) & ~(size_t)255;
        return p;
    };
    __bf16* ctx_bf   = (__bf16*)carve((size_t)B_ * C_ * 2);
    __bf16* Wh_bf    = (__bf16*)carve((size_t)H_ * C_ * 2);
    __bf16* Wc_bf    = (__bf16*)carve((size_t)H_ * C_ * 2);
    __bf16* Wih_bf   = (__bf16*)carve((size_t)HG_ * C_ * 2);
    __bf16* Whh_ext  = (__bf16*)carve((size_t)HG_ * HE_ * 2);
    __bf16* W1_bf    = (__bf16*)carve((size_t)H_ * H_ * 2);
    float*  gatesCtx = (float*) carve((size_t)B_ * HG_ * 4);
    float*  gatesBuf = (float*) carve((size_t)B_ * HG_ * 4);
    float*  cstate   = (float*) carve((size_t)B_ * H_ * 4);
    __bf16* h_ext    = (__bf16*)carve((size_t)B_ * HE_ * 2);
    __bf16* yln_bf   = (__bf16*)carve((size_t)B_ * H_ * 2);
    __bf16* y_bf     = (__bf16*)carve((size_t)B_ * H_ * 2);
    float*  posState = (float*) carve((size_t)B_ * 2 * 4);
    float*  deltaState = (float*)carve((size_t)B_ * 2 * 4);

    // ---- one-time precompute --------------------------------------------
    auto blks = [](long n) { return (int)((n + 255) / 256); };
    cvt_bf16_kernel<<<blks((long)B_ * C_), 256, 0, stream>>>(context, C_, ctx_bf, B_, C_);
    cvt_bf16_kernel<<<blks((long)H_ * C_), 256, 0, stream>>>(Wh, C_, Wh_bf, H_, C_);
    cvt_bf16_kernel<<<blks((long)H_ * C_), 256, 0, stream>>>(Wc, C_, Wc_bf, H_, C_);
    cvt_bf16_kernel<<<blks((long)HG_ * C_), 256, 0, stream>>>(W_ih, IN_, Wih_bf, HG_, C_);
    cvt_bf16_kernel<<<blks((long)H_ * H_), 256, 0, stream>>>(W1, H_, W1_bf, H_, H_);
    build_whh_ext_kernel<<<blks((long)HG_ * HE_), 256, 0, stream>>>(W_hh, W_ih, Whh_ext);
    init_state_kernel<<<blks(B_ * 2), 256, 0, stream>>>(last_pos, last_delta, posState, deltaState);

    // hidden = tanh(ctx @ Wh^T + bh) -> h_ext (bf16)
    gemm_bf16_wmma<<<dim3(H_ / BN, B_ / BM), 256, 0, stream>>>(
        ctx_bf, C_, Wh_bf, C_, bh, nullptr, nullptr, h_ext, HE_, H_, C_, 0);
    // cell = tanh(ctx @ Wc^T + bc) -> cstate (f32)
    gemm_bf16_wmma<<<dim3(H_ / BN, B_ / BM), 256, 0, stream>>>(
        ctx_bf, C_, Wc_bf, C_, bc, nullptr, cstate, nullptr, 0, H_, C_, 0);
    // gates_ctx = ctx @ W_ih[:, :C]^T + b_ih -> f32
    gemm_bf16_wmma<<<dim3(HG_ / BN, B_ / BM), 256, 0, stream>>>(
        ctx_bf, C_, Wih_bf, C_, b_ih, nullptr, gatesCtx, nullptr, 0, HG_, C_, 1);

    // ---- recurrent steps -------------------------------------------------
    for (int t = 0; t < T_; ++t) {
        fill_extra_kernel<<<B_ / 256, 256, 0, stream>>>(deltaState, h_ext);
        // gates = gates_ctx + h_ext @ Whh_ext^T + b_hh
        gemm_bf16_wmma<<<dim3(HG_ / BN, B_ / BM), 256, 0, stream>>>(
            h_ext, HE_, Whh_ext, HE_, b_hh, gatesCtx, gatesBuf, nullptr, 0, HG_, HE_, 2);
        lstm_pointwise_ln<<<B_, 256, 0, stream>>>(gatesBuf, cstate, h_ext, ln_g, ln_b, yln_bf);
        // y = gelu(yln @ W1^T + b1) -> bf16
        gemm_bf16_wmma<<<dim3(H_ / BN, B_ / BM), 256, 0, stream>>>(
            yln_bf, H_, W1_bf, H_, b1, nullptr, nullptr, y_bf, H_, H_, H_, 3);
        head_kernel<<<B_ / 8, 256, 0, stream>>>(y_bf, W2, b2, posState, deltaState, out, t);
    }
}
